// CartTensorOut_44057774522754
// MI455X (gfx1250) — matrix operand order repaired and checked by reading, not compiled, not run on hardware
//
#include <hip/hip_runtime.h>

// ---------------------------------------------------------------------------
// CDNA5 (gfx1250) fused equivariant TP kernel, round 2.
// v_wmma_f32_16x16x32_bf16 for all dense GEMMs; packed bf16 fragment builds;
// bounded unrolling to stay under 256 VGPRs (no MSB mode / no spills).
// ---------------------------------------------------------------------------

#define N_NODES 50000
#define NGRAPH  500
#define NODES_PER_BLOCK 128
#define THREADS 128
#define PROW 292   // padded per-node LDS row (bf16 elements) to reduce bank conflicts

typedef __bf16 bf16_t;
typedef __attribute__((ext_vector_type(16))) __bf16 v16bf;
typedef __attribute__((ext_vector_type(8)))  __bf16 v8bf;
typedef __attribute__((ext_vector_type(8)))  float  v8f;
typedef __attribute__((ext_vector_type(8)))  unsigned v8u;

// workspace layout (bf16 elements): transposed bf16 weights, B-matrix friendly [N][K]
#define OFF_MW1T 0          // [64][128]
#define OFF_MW2T 8192       // [288][64]
#define OFF_W0T  26624      // [32][128]
#define OFF_W1T  30720      // [32][64]
#define OFF_W2T  32768      // [32][32]

__device__ __forceinline__ bf16_t f2bf(float f) { return (bf16_t)f; }
__device__ __forceinline__ float  bf2f(bf16_t b) { return (float)b; }

// two floats -> one packed dword of bf16 (lo, hi)
__device__ __forceinline__ unsigned pack_bf2(float a, float b) {
  unsigned short x = __builtin_bit_cast(unsigned short, (bf16_t)a);
  unsigned short y = __builtin_bit_cast(unsigned short, (bf16_t)b);
  return (unsigned)x | ((unsigned)y << 16);
}

__device__ __forceinline__ v8f wmma_bf16(v16bf a, v16bf b, v8f c) {
  return __builtin_amdgcn_wmma_f32_16x16x32_bf16(false, a, false, b, (short)0, c,
                                                 false, false);
}

// B fragment (16-bit B 32x16): lane holds column N = lane&15; halves 0..7 -> K = kb..kb+7,
// halves 8..15 -> K = kb+16..kb+23, kb = 8*(lane>>4). Source is bf16 [N][K] row-major.
__device__ __forceinline__ v16bf load_b(const bf16_t* base, int ldk, int nt, int kt,
                                        int lo16, int hi) {
  const int n  = nt * 16 + lo16;
  const int kb = hi * 8 + kt * 32;
  const v8bf p0 = *reinterpret_cast<const v8bf*>(base + n * ldk + kb);
  const v8bf p1 = *reinterpret_cast<const v8bf*>(base + n * ldk + kb + 16);
  return __builtin_shufflevector(p0, p1, 0, 1, 2, 3, 4, 5, 6, 7,
                                 8, 9, 10, 11, 12, 13, 14, 15);
}

// A fragment (16-bit A 16x32): lane row M = lane&15; halves 0..7 -> K = kb..kb+7,
// halves 8..15 -> K = kb+16..kb+23, kb = 8*(lane>>4) + kt*32 (passed in).
__device__ __forceinline__ v16bf load_a_rowmajor_f32(const float* row, int kb) {
  float4 q0 = *reinterpret_cast<const float4*>(row + kb);
  float4 q1 = *reinterpret_cast<const float4*>(row + kb + 4);
  float4 q2 = *reinterpret_cast<const float4*>(row + kb + 16);
  float4 q3 = *reinterpret_cast<const float4*>(row + kb + 20);
  v8u t;
  t[0] = pack_bf2(q0.x, q0.y); t[1] = pack_bf2(q0.z, q0.w);
  t[2] = pack_bf2(q1.x, q1.y); t[3] = pack_bf2(q1.z, q1.w);
  t[4] = pack_bf2(q2.x, q2.y); t[5] = pack_bf2(q2.z, q2.w);
  t[6] = pack_bf2(q3.x, q3.y); t[7] = pack_bf2(q3.z, q3.w);
  return __builtin_bit_cast(v16bf, t);
}

// ---------------------------------------------------------------------------
// Init kernel: zero output accumulators, convert+transpose weights to bf16 ws.
// ---------------------------------------------------------------------------
__global__ void init_kernel(const float* __restrict__ W0, const float* __restrict__ W1,
                            const float* __restrict__ W2, const float* __restrict__ mw1,
                            const float* __restrict__ mw2, float* __restrict__ out,
                            bf16_t* __restrict__ ws) {
  const int i0 = blockIdx.x * blockDim.x + threadIdx.x;
  const int st = gridDim.x * blockDim.x;
  for (int k = i0; k < NGRAPH * 9; k += st) out[k] = 0.0f;
  for (int k = i0; k < 64 * 128; k += st) {            // mw1T[o][i] = mw1[i][o]
    int o = k >> 7, in = k & 127;
    ws[OFF_MW1T + k] = f2bf(mw1[in * 64 + o]);
  }
  for (int k = i0; k < 288 * 64; k += st) {            // mw2T[o][i] = mw2[i][o]
    int o = k >> 6, in = k & 63;
    ws[OFF_MW2T + k] = f2bf(mw2[in * 288 + o]);
  }
  for (int k = i0; k < 32 * 128; k += st) {            // W0T[c][u]
    int c = k >> 7, u = k & 127;
    ws[OFF_W0T + k] = f2bf(W0[u * 32 + c]);
  }
  for (int k = i0; k < 32 * 64; k += st) {             // W1T[c][u]
    int c = k >> 6, u = k & 63;
    ws[OFF_W1T + k] = f2bf(W1[u * 32 + c]);
  }
  for (int k = i0; k < 32 * 32; k += st) {             // W2T[c][u]
    int c = k >> 5, u = k & 31;
    ws[OFF_W2T + k] = f2bf(W2[u * 32 + c]);
  }
}

// ---------------------------------------------------------------------------
// Fused main kernel: 4 waves, 128 nodes / block.
// ---------------------------------------------------------------------------
__global__ __launch_bounds__(THREADS, 1)
void fused_kernel(const int* __restrict__ batch,
                  const float* __restrict__ node_scalar,
                  const float* __restrict__ node_equi,
                  const float* __restrict__ mb1,
                  const float* __restrict__ mb2,
                  const bf16_t* __restrict__ ws,
                  float* __restrict__ out) {
  __shared__ float  sMb1[64];
  __shared__ float  sMb2[288];
  __shared__ __align__(16) bf16_t sH[4][16 * 64];              // per-wave silu(h) tile
  __shared__ __align__(16) bf16_t sW[NODES_PER_BLOCK * PROW];  // TP weights  [node][c*9+i]
  __shared__ __align__(16) bf16_t sY[NODES_PER_BLOCK * PROW];  // raw z       [node][...]

  const int tid  = threadIdx.x;
  const int wid  = tid >> 5;
  const int lane = tid & 31;
  const int lo16 = lane & 15;
  const int hi   = lane >> 4;
  const v8f vzero = {0.f, 0.f, 0.f, 0.f, 0.f, 0.f, 0.f, 0.f};

  for (int k = tid; k < 64;  k += THREADS) sMb1[k] = mb1[k];
  for (int k = tid; k < 288; k += THREADS) sMb2[k] = mb2[k];
  __syncthreads();

  const bf16_t* wMw1T = ws + OFF_MW1T;
  const bf16_t* wMw2T = ws + OFF_MW2T;
  const bf16_t* wW0T  = ws + OFF_W0T;
  const bf16_t* wW1T  = ws + OFF_W1T;
  const bf16_t* wW2T  = ws + OFF_W2T;

  for (int mt = 0; mt < 2; ++mt) {
    const int tbase_l = wid * 32 + mt * 16;
    int nrow = blockIdx.x * NODES_PER_BLOCK + tbase_l + lo16;
    if (nrow >= N_NODES) nrow = N_NODES - 1;                 // clamp (uniform EXEC)
    const float* srow = node_scalar + (size_t)nrow * 128;
    const float* erow = node_equi   + (size_t)nrow * 448;
    __builtin_prefetch(erow, 0, 1);                          // global_prefetch_b8
    __builtin_prefetch(erow + 256, 0, 1);

    // ---- GEMM1: H = silu(S[16,128] @ mw1 + mb1) -> sH (bf16) -------------
    v16bf aS[4];
#pragma unroll
    for (int kt = 0; kt < 4; ++kt)
      aS[kt] = load_a_rowmajor_f32(srow, hi * 8 + kt * 32);
#pragma unroll 2
    for (int nt = 0; nt < 4; ++nt) {
      v8f acc = vzero;
#pragma unroll
      for (int kt = 0; kt < 4; ++kt)
        acc = wmma_bf16(aS[kt], load_b(wMw1T, 128, nt, kt, lo16, hi), acc);
      const int hd = nt * 16 + lo16;
      const float bias = sMb1[hd];
#pragma unroll
      for (int r = 0; r < 8; ++r) {
        float v = acc[r] + bias;
        v = v / (1.0f + __expf(-v));                         // silu
        sH[wid][(r + 8 * hi) * 64 + hd] = f2bf(v);
      }
    }

    // ---- GEMM2: W = H[16,64] @ mw2 + mb2 -> sW (bf16, [node][c*9+i]) -----
    v16bf aH[2];
#pragma unroll
    for (int kt = 0; kt < 2; ++kt) {
      const int kb = hi * 8 + kt * 32;
      const v8bf p0 = *reinterpret_cast<const v8bf*>(&sH[wid][lo16 * 64 + kb]);
      const v8bf p1 = *reinterpret_cast<const v8bf*>(&sH[wid][lo16 * 64 + kb + 16]);
      aH[kt] = __builtin_shufflevector(p0, p1, 0, 1, 2, 3, 4, 5, 6, 7,
                                       8, 9, 10, 11, 12, 13, 14, 15);
    }
#pragma unroll 2
    for (int nt = 0; nt < 18; ++nt) {
      v8f acc = vzero;
#pragma unroll
      for (int kt = 0; kt < 2; ++kt)
        acc = wmma_bf16(aH[kt], load_b(wMw2T, 64, nt, kt, lo16, hi), acc);
      const int n2 = nt * 16 + lo16;
      const float bias = sMb2[n2];
      const int fi = n2 >> 5, fc = n2 & 31;                  // instr, channel
#pragma unroll
      for (int r = 0; r < 8; ++r)
        sW[(tbase_l + r + 8 * hi) * PROW + fc * 9 + fi] = f2bf(acc[r] + bias);
    }

    // ---- channel mix l=0: z0 = X0[16,128] @ W0 ---------------------------
    v16bf a0[4];
#pragma unroll
    for (int kt = 0; kt < 4; ++kt)
      a0[kt] = load_a_rowmajor_f32(erow, hi * 8 + kt * 32);
#pragma unroll 2
    for (int nt = 0; nt < 2; ++nt) {
      v8f acc = vzero;
#pragma unroll
      for (int kt = 0; kt < 4; ++kt)
        acc = wmma_bf16(a0[kt], load_b(wW0T, 128, nt, kt, lo16, hi), acc);
      const int c = nt * 16 + lo16;
#pragma unroll
      for (int r = 0; r < 8; ++r)
        sY[(tbase_l + r + 8 * hi) * PROW + c] = f2bf(acc[r]);
    }

    // ---- channel mix l=1 (3 GEMMs share B): z1[m] = X1m[16,64] @ W1 ------
    // x1[u][m] at erow + 128 + u*3 + m; pack u-pairs into bf16 dwords.
    v8u a1u[3][2];
#pragma unroll
    for (int kt = 0; kt < 2; ++kt) {
      const int kb = hi * 8 + kt * 32;
#pragma unroll
      for (int h2 = 0; h2 < 4; ++h2) {
        const float* pA = erow + 128 + (kb + 2 * h2) * 3;        // u, u+1
        const float* pB = erow + 128 + (kb + 16 + 2 * h2) * 3;
#pragma unroll
        for (int m = 0; m < 3; ++m) {
          a1u[m][kt][h2]     = pack_bf2(pA[m], pA[m + 3]);
          a1u[m][kt][4 + h2] = pack_bf2(pB[m], pB[m + 3]);
        }
      }
    }
#pragma unroll 1
    for (int nt = 0; nt < 2; ++nt) {
      v8f am0 = vzero, am1 = vzero, am2 = vzero;
#pragma unroll
      for (int kt = 0; kt < 2; ++kt) {
        const v16bf b = load_b(wW1T, 64, nt, kt, lo16, hi);
        am0 = wmma_bf16(__builtin_bit_cast(v16bf, a1u[0][kt]), b, am0);
        am1 = wmma_bf16(__builtin_bit_cast(v16bf, a1u[1][kt]), b, am1);
        am2 = wmma_bf16(__builtin_bit_cast(v16bf, a1u[2][kt]), b, am2);
      }
      const int c = nt * 16 + lo16;
#pragma unroll
      for (int r = 0; r < 8; ++r) {
        const int nl = tbase_l + r + 8 * hi;
        sY[nl * PROW + 32 + c * 3 + 0] = f2bf(am0[r]);
        sY[nl * PROW + 32 + c * 3 + 1] = f2bf(am1[r]);
        sY[nl * PROW + 32 + c * 3 + 2] = f2bf(am2[r]);
      }
    }

    // ---- channel mix l=2 (5 GEMMs share B): z2[m] = X2m[16,32] @ W2 ------
    // x2[u][m] at erow + 320 + u*5 + m
    v8u a2u[5];
    {
      const int kb = hi * 8;
#pragma unroll
      for (int h2 = 0; h2 < 4; ++h2) {
        const float* pA = erow + 320 + (kb + 2 * h2) * 5;
        const float* pB = erow + 320 + (kb + 16 + 2 * h2) * 5;
#pragma unroll
        for (int m = 0; m < 5; ++m) {
          a2u[m][h2]     = pack_bf2(pA[m], pA[m + 5]);
          a2u[m][4 + h2] = pack_bf2(pB[m], pB[m + 5]);
        }
      }
    }
#pragma unroll 1
    for (int nt = 0; nt < 2; ++nt) {
      const v16bf b = load_b(wW2T, 32, nt, 0, lo16, hi);
      v8f am[5];
#pragma unroll
      for (int m = 0; m < 5; ++m)
        am[m] = wmma_bf16(__builtin_bit_cast(v16bf, a2u[m]), b, vzero);
      const int c = nt * 16 + lo16;
#pragma unroll
      for (int r = 0; r < 8; ++r) {
        const int nl = tbase_l + r + 8 * hi;
#pragma unroll
        for (int m = 0; m < 5; ++m)
          sY[nl * PROW + 128 + c * 5 + m] = f2bf(am[m][r]);
      }
    }
  }

  __syncthreads();

  // ---- per-node: RMS norm + sparse CG TP + Cartesian map + segment sum ---
  const int node = blockIdx.x * NODES_PER_BLOCK + tid;
  if (node < N_NODES) {
    const bf16_t* y = &sY[tid * PROW];
    const bf16_t* w = &sW[tid * PROW];

    // pass 1: RMS over channels (sum over m of z^2, mean over 32 channels)
    float s0 = 0.f, s1 = 0.f, s2 = 0.f;
    for (int c = 0; c < 32; ++c)  { float z = bf2f(y[c]);        s0 += z * z; }
    for (int k = 0; k < 96; ++k)  { float z = bf2f(y[32 + k]);   s1 += z * z; }
    for (int k = 0; k < 160; ++k) { float z = bf2f(y[128 + k]);  s2 += z * z; }
    const float inv0 = rsqrtf(s0 * 0.03125f + 1e-5f);
    const float inv1 = rsqrtf(s1 * 0.03125f + 1e-5f);
    const float inv2 = rsqrtf(s2 * 0.03125f + 1e-5f);

    // Real-CG constants (basis order l=1: y,z,x ; l=2: xy,yz,z2,xz,x2-y2)
    const float C110 = -0.57735026919f;   // -1/sqrt(3)
    const float C220 =  0.44721359550f;   //  1/sqrt(5)
    const float C022 =  0.44721359550f;
    const float A1 = 0.63245553203f;      // 2/sqrt(10)
    const float A2 = 0.18257418584f;      // 1/sqrt(30)
    const float A3 = 0.31622776602f;      // 1/sqrt(10)
    const float B1 = 0.47809144374f;      // 4/sqrt(70)
    const float B2 = 0.41403933560f;      // 2*sqrt(3/70)
    const float B3 = 0.23904572187f;      // 2/sqrt(70)
    const float B4 = 0.11952286093f;      // 1/sqrt(70)
    const float B5 = 0.20701966780f;      // sqrt(3/70)

    float o0 = 0.f, o2a = 0.f, o2b = 0.f, o2c = 0.f, o2d = 0.f, o2e = 0.f;
    // (1,1,1) and (2,2,1) couplings are antisymmetric with identical inputs
    // -> contribute exactly zero -> out_l1 == 0.
    for (int c = 0; c < 32; ++c) {
      const float y0 = bf2f(y[c]) * inv0;
      const float t0 = bf2f(y[32 + c * 3 + 0]) * inv1;   // y
      const float t1 = bf2f(y[32 + c * 3 + 1]) * inv1;   // z
      const float t2 = bf2f(y[32 + c * 3 + 2]) * inv1;   // x
      const float u0 = bf2f(y[128 + c * 5 + 0]) * inv2;
      const float u1 = bf2f(y[128 + c * 5 + 1]) * inv2;
      const float u2 = bf2f(y[128 + c * 5 + 2]) * inv2;
      const float u3 = bf2f(y[128 + c * 5 + 3]) * inv2;
      const float u4 = bf2f(y[128 + c * 5 + 4]) * inv2;
      const float w0 = bf2f(w[c * 9 + 0]);
      const float w1 = bf2f(w[c * 9 + 1]);
      const float w2 = bf2f(w[c * 9 + 2]);
      const float w5 = bf2f(w[c * 9 + 5]);
      const float w6 = bf2f(w[c * 9 + 6]);
      const float w7 = bf2f(w[c * 9 + 7]);
      const float w8 = bf2f(w[c * 9 + 8]);

      o0 += w0 * y0 * y0;
      o0 += C110 * w1 * (t0 * t0 + t1 * t1 + t2 * t2);
      o0 += C220 * w2 * (u0*u0 + u1*u1 + u2*u2 + u3*u3 + u4*u4);

      // (1,1,2) quadrupole of y1
      const float q0 = A1 * t2 * t0;
      const float q1 = A1 * t0 * t1;
      const float q2 = A2 * (2.f * t1 * t1 - t0 * t0 - t2 * t2);
      const float q3 = A1 * t2 * t1;
      const float q4 = A3 * (t2 * t2 - t0 * t0);
      // (2,2,2) symmetric coupling of y2
      const float p0 = B1 * u0 * u2 - B2 * u1 * u3;
      const float p1 = B2 * (u1 * u4 - u0 * u3) - B3 * u1 * u2;
      const float p2 = B4 * (2.f*u0*u0 - u1*u1 - 2.f*u2*u2 - u3*u3 + 2.f*u4*u4);
      const float p3 = -B2 * (u3 * u4 + u0 * u1) - B3 * u2 * u3;
      const float p4 = B1 * u2 * u4 + B5 * (u1 * u1 - u3 * u3);
      const float g78 = C022 * y0 * (w7 + w8);     // (0,2,2)+(2,0,2)

      o2a += w5 * q0 + w6 * p0 + g78 * u0;
      o2b += w5 * q1 + w6 * p1 + g78 * u1;
      o2c += w5 * q2 + w6 * p2 + g78 * u2;
      o2d += w5 * q3 + w6 * p3 + g78 * u3;
      o2e += w5 * q4 + w6 * p4 + g78 * u4;
    }
    o0  *= 0.10206207262f;                 // 1/sqrt(C*3)
    o2a *= 0.08838834765f; o2b *= 0.08838834765f; o2c *= 0.08838834765f;  // 1/sqrt(C*4)
    o2d *= 0.08838834765f; o2e *= 0.08838834765f;

    // cart[i][j] = Q[i,j,k] sph[k]  (sph = [o0, 0,0,0, o2a..o2e])
    const float diag = C110 * o0;
    float cart[3][3];
    cart[0][0] = diag - A2 * o2c - A3 * o2e;     // yy
    cart[1][1] = diag + 2.f * A2 * o2c;          // zz
    cart[2][2] = diag - A2 * o2c + A3 * o2e;     // xx
    cart[0][1] = cart[1][0] = A3 * o2b;          // yz
    cart[1][2] = cart[2][1] = A3 * o2d;          // zx
    cart[0][2] = cart[2][0] = A3 * o2a;          // yx

    const int g = batch[node];
    float* og = out + g * 9;
#pragma unroll
    for (int a = 0; a < 3; ++a)
#pragma unroll
      for (int b = 0; b < 3; ++b)                // fold roll(+1,+1) into write index
        atomicAdd(&og[((a + 1) % 3) * 3 + ((b + 1) % 3)], cart[a][b]);
  }
}

// ---------------------------------------------------------------------------
extern "C" void kernel_launch(void* const* d_in, const int* in_sizes, int n_in,
                              void* d_out, int out_size, void* d_ws, size_t ws_size,
                              hipStream_t stream) {
  (void)in_sizes; (void)n_in; (void)out_size; (void)ws_size;
  const int*   batch = (const int*)  d_in[0];
  const float* ns    = (const float*)d_in[1];
  const float* ne    = (const float*)d_in[2];
  const float* W0    = (const float*)d_in[3];
  const float* W1    = (const float*)d_in[4];
  const float* W2    = (const float*)d_in[5];
  const float* mw1   = (const float*)d_in[6];
  const float* mb1   = (const float*)d_in[7];
  const float* mw2   = (const float*)d_in[8];
  const float* mb2   = (const float*)d_in[9];
  float*  out = (float*)d_out;
  bf16_t* ws  = (bf16_t*)d_ws;

  init_kernel<<<64, 256, 0, stream>>>(W0, W1, W2, mw1, mw2, out, ws);
  const int nblk = (N_NODES + NODES_PER_BLOCK - 1) / NODES_PER_BLOCK;
  fused_kernel<<<nblk, THREADS, 0, stream>>>(batch, ns, ne, mb1, mb2, ws, out);
}
